// MoE_10041633538672
// MI455X (gfx1250) — compile-verified, matrix operand order checked by reading, not run on hardware
//
#include <hip/hip_runtime.h>
#include <hip/hip_bf16.h>

#define S_DIM 8192
#define D_DIM 2048
#define H_DIM 64
#define E_DIM 64
#define L_DIM 64
#define LD_SZ (L_DIM * D_DIM)     // 131072 elements per expert

// LDS B-tile: 128 rows (2 experts x 64 L) x 32 K bf16 = 64B/row, padded +48B -> 112B stride
#define BROWS   128
#define BROWB   112               // 64B data + 48B pad (TDM pad_enable)
#define BBUFB   (BROWS * BROWB)   // 14336 B per buffer

typedef __attribute__((ext_vector_type(16))) __bf16 v16bf;
typedef __attribute__((ext_vector_type(8)))  float  v8f;
typedef unsigned int u32x4 __attribute__((ext_vector_type(4)));
typedef int          i32x8 __attribute__((ext_vector_type(8)));
typedef int          i32x4 __attribute__((ext_vector_type(4)));

union FragBF {
    uint4 q[2];   // 2 x 16B = 32B = 16 bf16
    v16bf v;
};

// ---------------------------------------------------------------------------
// Kernel 1: convert x -> bf16 AND per-block partials of u[d] = sum_s wout[s]*x[s,d]
// (gate collapsed algebraically: scores = ((wout @ x) @ Wg_in^T) @ Wg_lin^T)
// ---------------------------------------------------------------------------
__global__ __launch_bounds__(256) void convert_gate_partial(
    const float* __restrict__ x, const float* __restrict__ wout,
    __hip_bfloat16* __restrict__ xb, float* __restrict__ partial)
{
    const int b = blockIdx.x;
    const int t = threadIdx.x;
    float acc[8] = {0.f, 0.f, 0.f, 0.f, 0.f, 0.f, 0.f, 0.f};
    const int row0 = b * 64;
    for (int s = 0; s < 64; ++s) {
        const int r = row0 + s;
        const float w = wout[r];
        const float* __restrict__ xr = x + (size_t)r * D_DIM;
        __hip_bfloat16* __restrict__ xbr = xb + (size_t)r * D_DIM;
        #pragma unroll
        for (int j = 0; j < 8; ++j) {
            const int d = t + j * 256;
            const float vv = xr[d];
            xbr[d] = __float2bfloat16(vv);
            acc[j] += vv * w;
        }
    }
    #pragma unroll
    for (int j = 0; j < 8; ++j)
        partial[(size_t)b * D_DIM + t + j * 256] = acc[j];
}

__global__ __launch_bounds__(256) void reduce_u(
    const float* __restrict__ partial, float* __restrict__ u)
{
    const int d = blockIdx.x * 256 + threadIdx.x;   // 8 x 256 = 2048
    float s = 0.f;
    for (int b = 0; b < 128; ++b) s += partial[(size_t)b * D_DIM + d];
    u[d] = s;
}

__global__ __launch_bounds__(256) void gate_v_kernel(
    const float* __restrict__ u, const float* __restrict__ Wg_in,
    float* __restrict__ v)
{
    __shared__ float red[256];
    const int h = blockIdx.x;
    const float* __restrict__ wr = Wg_in + (size_t)h * D_DIM;
    float a = 0.f;
    for (int d = threadIdx.x; d < D_DIM; d += 256) a += u[d] * wr[d];
    red[threadIdx.x] = a;
    __syncthreads();
    for (int s = 128; s > 0; s >>= 1) {
        if (threadIdx.x < s) red[threadIdx.x] += red[threadIdx.x + s];
        __syncthreads();
    }
    if (threadIdx.x == 0) v[h] = red[0];
}

__global__ void gate_topk(const float* __restrict__ v, const float* __restrict__ Wg_lin,
                          float* __restrict__ g, int* __restrict__ idx)
{
    __shared__ float sc[E_DIM];
    const int e = threadIdx.x;   // 64 threads
    float s = 0.f;
    #pragma unroll 8
    for (int h = 0; h < H_DIM; ++h) s += v[h] * Wg_lin[e * H_DIM + h];
    sc[e] = s;
    __syncthreads();
    if (e == 0) {
        int i0 = 0; float v0 = sc[0];
        for (int i = 1; i < E_DIM; ++i) if (sc[i] > v0) { v0 = sc[i]; i0 = i; }
        int i1 = (i0 == 0) ? 1 : 0; float v1 = sc[i1];
        for (int i = 0; i < E_DIM; ++i)
            if (i != i0 && sc[i] > v1) { v1 = sc[i]; i1 = i; }
        const float e1  = __expf(v1 - v0);
        const float inv = 1.0f / (1.0f + e1);
        g[0] = inv;
        g[1] = e1 * inv;
        idx[0] = i0;
        idx[1] = i1;
    }
}

__global__ __launch_bounds__(256) void gather_experts(
    const float* __restrict__ We, const int* __restrict__ idx,
    __hip_bfloat16* __restrict__ wsel)
{
    const int total = 2 * LD_SZ;
    for (int i = blockIdx.x * blockDim.x + threadIdx.x; i < total;
         i += gridDim.x * blockDim.x) {
        const int k = (i >= LD_SZ) ? 1 : 0;
        const int off = i - k * LD_SZ;
        wsel[i] = __float2bfloat16(We[(size_t)idx[k] * LD_SZ + off]);
    }
}

// ---------------------------------------------------------------------------
// TDM issue: load one K-step B tile (128 rows x 32 bf16, tensor row stride 2048)
// into LDS at ldsOff, padding each 64B row with 48B (-> 112B LDS row stride).
// D# groups per CDNA5 ISA ch.8; clang-23 6-arg builtin form.
// ---------------------------------------------------------------------------
__device__ __forceinline__ void tdm_issue_btile(const __hip_bfloat16* gsrc,
                                                unsigned int ldsOff)
{
    const unsigned long long ga = (unsigned long long)(uintptr_t)gsrc;

    u32x4 g0;
    g0[0] = 1u;                                   // count=1, is_restore=0, gather off
    g0[1] = ldsOff;                               // lds_addr (bytes)
    g0[2] = (unsigned int)(ga & 0xFFFFFFFFu);     // global_addr[31:0]
    g0[3] = (unsigned int)(ga >> 32) | (2u << 30);// global_addr[56:32] | type=2 (image)

    i32x8 g1;
    g1[0] = (int)((1u << 16)    // data_size = 1 -> 2 bytes
                | (1u << 20)    // pad_enable
                | (3u << 22)    // pad_interval: 16 dwords (64B)
                | (11u << 25)); // pad_amount: 12 dwords (48B)
    g1[1] = (int)(2048u << 16); // tensor_dim0 = 2048 (low16 at bits 63:48)
    g1[2] = (int)(128u << 16);  // tensor_dim0 hi=0 | tensor_dim1 = 128 (low16)
    g1[3] = (int)(32u << 16);   // tensor_dim1 hi=0 | tile_dim0 = 32
    g1[4] = 128;                // tile_dim1 = 128 | tile_dim2 = 0
    g1[5] = 2048;               // tensor_dim0_stride low32 (elements)
    g1[6] = 0;                  // stride0 hi | tensor_dim1_stride low16
    g1[7] = 0;                  // tensor_dim1_stride hi

    i32x4 g2 = {1, 0, 0, 0};    // tensor_dim2 = 1; rest unused (2D tile)
    i32x4 g3 = {0, 0, 0, 0};
    i32x8 g4 = {0, 0, 0, 0, 0, 0, 0, 0};   // unused trailing group (clang-23 form)

    __builtin_amdgcn_tensor_load_to_lds(g0, g1, g2, g3, g4, 0);
}

// ---------------------------------------------------------------------------
// Kernel 6: fused dual-expert GEMM (bf16 WMMA, f32 acc) + L2-normalize + exact
// GELU + gate-weighted combine. B staged per K-step into LDS by the Tensor
// Data Mover (double buffered, s_wait_tensorcnt + barrier), shared by 8 waves.
// A frags stream from global (each x row read exactly once device-wide).
// ---------------------------------------------------------------------------
__global__ __launch_bounds__(256) void expert_wmma(
    const __hip_bfloat16* __restrict__ xb,
    const __hip_bfloat16* __restrict__ wsel,
    const float* __restrict__ gbuf,
    float* __restrict__ out)
{
    __shared__ __align__(16) unsigned char Bs[2][BBUFB];   // 2 x 14KB

    const int lane    = threadIdx.x & 31;
    const int wave    = threadIdx.x >> 5;
    const int rowBase = blockIdx.x * 128 + wave * 16;
    const int mrow    = lane & 15;
    const int khalf   = (lane & 16) ? 8 : 0;     // K sub-half selector (ISA 16-bit layout)

    v8f acc[2][4];
    const v8f vzero = {};
    #pragma unroll
    for (int k = 0; k < 2; ++k)
        #pragma unroll
        for (int t = 0; t < 4; ++t) acc[k][t] = vzero;

    const __hip_bfloat16* __restrict__ arowp =
        xb + (size_t)(rowBase + mrow) * D_DIM + khalf;

    const unsigned int bsOff[2] = {
        (unsigned int)(uintptr_t)&Bs[0][0],
        (unsigned int)(uintptr_t)&Bs[1][0]
    };

    // Prologue: TDM-load K-step 0 into buffer 0
    if (wave == 0) tdm_issue_btile(wsel, bsOff[0]);

    for (int step = 0; step < D_DIM / 32; ++step) {
        const int cur = step & 1;
        if (wave == 0) __builtin_amdgcn_s_wait_tensorcnt(0);  // buf[cur] landed
        __syncthreads();  // releases waves; also proves buf[cur^1] reads (step-1) done
        if (wave == 0 && step + 1 < D_DIM / 32)
            tdm_issue_btile(wsel + (size_t)(step + 1) * 32, bsOff[cur ^ 1]);

        const int d0 = step * 32;
        FragBF a;
        a.q[0] = *reinterpret_cast<const uint4*>(arowp + d0);       // K d0+{0..7 | 8..15}
        a.q[1] = *reinterpret_cast<const uint4*>(arowp + d0 + 16);  // K d0+{16..23 | 24..31}

        const unsigned char* __restrict__ bbase = &Bs[cur][0];
        #pragma unroll
        for (int k = 0; k < 2; ++k) {
            #pragma unroll
            for (int t = 0; t < 4; ++t) {
                const unsigned char* bp =
                    bbase + (unsigned)(k * 64 + t * 16 + mrow) * BROWB + khalf * 2;
                FragBF b;
                b.q[0] = *reinterpret_cast<const uint4*>(bp);        // K half lo
                b.q[1] = *reinterpret_cast<const uint4*>(bp + 32);   // K half hi
                acc[k][t] = __builtin_amdgcn_wmma_f32_16x16x32_bf16(
                    false, a.v, false, b.v, (short)0, acc[k][t], false, false);
            }
        }
    }

    const float g0 = gbuf[0];
    const float g1 = gbuf[1];

    float outv[4][8];
    #pragma unroll
    for (int t = 0; t < 4; ++t)
        #pragma unroll
        for (int v = 0; v < 8; ++v) outv[t][v] = 0.f;

    #pragma unroll
    for (int k = 0; k < 2; ++k) {
        const float gk = (k == 0) ? g0 : g1;
        #pragma unroll
        for (int v = 0; v < 8; ++v) {
            float ss = 0.f;
            #pragma unroll
            for (int t = 0; t < 4; ++t) { const float e = acc[k][t][v]; ss += e * e; }
            ss += __shfl_xor(ss, 1, 32);
            ss += __shfl_xor(ss, 2, 32);
            ss += __shfl_xor(ss, 4, 32);
            ss += __shfl_xor(ss, 8, 32);
            const float scale = 1.0f / fmaxf(sqrtf(ss), 1e-12f);
            #pragma unroll
            for (int t = 0; t < 4; ++t) {
                const float z  = acc[k][t][v] * scale;
                const float ge = 0.5f * z * (1.0f + erff(z * 0.70710678118654752f));
                outv[t][v] += gk * ge;
            }
        }
    }

    const int rOut = rowBase + ((lane & 16) ? 8 : 0);
    #pragma unroll
    for (int v = 0; v < 8; ++v) {
        float* __restrict__ orow = out + (size_t)(rOut + v) * L_DIM + mrow;
        #pragma unroll
        for (int t = 0; t < 4; ++t) orow[t * 16] = outv[t][v];
    }
}

// ---------------------------------------------------------------------------
extern "C" void kernel_launch(void* const* d_in, const int* in_sizes, int n_in,
                              void* d_out, int out_size, void* d_ws, size_t ws_size,
                              hipStream_t stream) {
    (void)in_sizes; (void)n_in; (void)out_size; (void)ws_size;
    const float* x      = (const float*)d_in[0];   // [S, D]
    const float* Wg_in  = (const float*)d_in[1];   // [H, D]
    const float* Wg_lin = (const float*)d_in[2];   // [E, H]
    const float* Wg_out = (const float*)d_in[3];   // [1, S]
    const float* We     = (const float*)d_in[4];   // [E, L, D]
    float* out = (float*)d_out;                    // [S, L] f32

    char* ws = (char*)d_ws;
    __hip_bfloat16* xb    = (__hip_bfloat16*)(ws);              // S*D bf16   = 32 MB
    __hip_bfloat16* wselb = (__hip_bfloat16*)(ws + 33554432);   // 2*L*D bf16 = 512 KB
    float* partial        = (float*)(ws + 34078720);            // 128*2048 f32 = 1 MB
    float* u              = (float*)(ws + 35127296);            // 2048 f32
    float* v              = (float*)(ws + 35135488);            // 64 f32
    float* g              = (float*)(ws + 35135744);            // 2 f32
    int*   idx            = (int*)  (ws + 35135752);            // 2 i32

    convert_gate_partial<<<128, 256, 0, stream>>>(x, Wg_out, xb, partial);
    reduce_u<<<8, 256, 0, stream>>>(partial, u);
    gate_v_kernel<<<64, 256, 0, stream>>>(u, Wg_in, v);
    gate_topk<<<1, 64, 0, stream>>>(v, Wg_lin, g, idx);
    gather_experts<<<256, 256, 0, stream>>>(We, idx, wselb);
    expert_wmma<<<64, 256, 0, stream>>>(xb, wselb, g, out);
}
